// VC_MODEL_10084583211048
// MI455X (gfx1250) — matrix operand order repaired and checked by reading, not compiled, not run on hardware
//
#include <hip/hip_runtime.h>
#include <hip/hip_bf16.h>
#include <math.h>

// ---------------- model dims ----------------
#define BB   512
#define INCH 240
#define TT   240
#define CH   128
#define DIM  256
#define NEMB 128
#define L0   120     // encoder length after stride-2 conv
#define L3   122     // length after conv3 (k=1, pad=1)
#define NPAD 128     // padded N for 16-wide WMMA tiles

typedef __attribute__((ext_vector_type(16))) _Float16 v16h;
typedef __attribute__((ext_vector_type(8)))  float    v8f;

// =====================================================================
// Generic f32 -> f16 cast (weights)
// =====================================================================
__global__ void cast_f16_kernel(const float* __restrict__ s, _Float16* __restrict__ d, int n) {
    int i = blockIdx.x * 256 + threadIdx.x;
    if (i < n) d[i] = (_Float16)s[i];
}

// =====================================================================
// Normalized codebook:  en[n][d] = embed_w[n][d] / ||embed_w[n]||   (f16)
// =====================================================================
__global__ __launch_bounds__(256) void embed_norm_kernel(const float* __restrict__ ew,
                                                         _Float16* __restrict__ en) {
    __shared__ float red[256];
    int n = blockIdx.x, d = threadIdx.x;
    float v = ew[n * DIM + d];
    red[d] = v * v; __syncthreads();
    for (int s = 128; s > 0; s >>= 1) { if (d < s) red[d] += red[d + s]; __syncthreads(); }
    float inv = 1.f / sqrtf(red[0]);
    en[n * DIM + d] = (_Float16)(v * inv);
}

// =====================================================================
// Fused im2col + WMMA GEMM (1-D conv as GEMM, no col buffer).
// Templated conv geometry (division folds to shift/mulhi); B-staging is
// branch-free: indices are clamped so the gather load is unconditional
// (legal address), then zero-selected -> no saveexec chains, loads clause up.
//   Cout[b][m][l] = act( sum_{ci,kw} A[m][ci*KW+kw] * src[b][ci][l*STRIDE+kw-PAD] + bias[m] )
// A: f16 [M][K=Cin*KW]; src: f32 [B][Cin][LinStride]; Cout: f32 [B][M][128]
// block = 256 threads = 8 wave32s; each wave owns one 16x16 C tile of a
// 128x16 macro-tile; K staged 64-deep in LDS => 2 v_wmma per barrier pair.
// grid = (8 ntiles, M/128, B).  ACT: 0 none, 1 relu, 2 4*relu (fused resblocks)
// =====================================================================
template <int KW, int STRIDE, int PAD, int ACT>
__global__ __launch_bounds__(256) void conv_gemm_wmma_kernel(
    const _Float16* __restrict__ A,
    const float* __restrict__ src,
    float* __restrict__ Cout,
    const float* __restrict__ bias,
    int M, int Cin, int Lin, int LinStride, int Lout)
{
    __shared__ _Float16 As[128 * 64];   // 16 KB : A chunk [128 rows][64 k]
    __shared__ _Float16 Bs[16 * 64];    //  2 KB : B chunk transposed [16 n][64 k]

    const int b     = blockIdx.z;
    const int mbase = blockIdx.y * 128;
    const int ntile = blockIdx.x;
    const int tid   = threadIdx.x;
    const int lane  = tid & 31;
    const int wave  = tid >> 5;
    const int lrow  = lane & 15;
    const int khalf = lane >> 4;
    const int K     = Cin * KW;

    // A staging: each thread copies 32 contiguous halfs (64 B)
    const int arow = tid >> 1;
    const int acol = (tid & 1) * 32;
    // B staging: n fastest across threads (coalesced src reads), 4 k's per thread
    const int bn    = tid & 15;
    const int bk0   = (tid >> 4) * 4;
    const int nglob = ntile * 16 + bn;           // output position l
    const bool nok  = true;                      // folded below per element
    const float* srcb = src + (long)b * Cin * LinStride;

    v8f acc = {};
    for (int kk = 0; kk < K; kk += 64) {
        // ---- stage A ----
        const _Float16* ag = A + (long)(mbase + arow) * K + kk + acol;
        __builtin_prefetch((const void*)(ag + 64), 0, 1);   // global_prefetch_b8
        _Float16* as = &As[arow * 64 + acol];
        #pragma unroll
        for (int j = 0; j < 32; ++j) as[j] = ag[j];
        // ---- stage B: branch-free fused im2col gather + f32->f16 ----
        #pragma unroll
        for (int j = 0; j < 4; ++j) {
            int k  = kk + bk0 + j;
            int ci = k / KW;                 // constant divisor -> shift / mulhi
            int kw = k - ci * KW;
            int s  = nglob * STRIDE + kw - PAD;
            int sc = s < 0 ? 0 : (s >= Lin ? Lin - 1 : s);   // clamp -> legal addr
            float v = srcb[(long)ci * LinStride + sc];       // unconditional load
            bool ok = (nglob < Lout) & (s >= 0) & (s < Lin);
            Bs[bn * 64 + bk0 + j] = (_Float16)(ok ? v : 0.f); // cndmask select
        }
        __syncthreads();

        const _Float16* ap = &As[(wave * 16 + lrow) * 64];
        const _Float16* bp = &Bs[lrow * 64];
        #pragma unroll
        for (int sub = 0; sub < 2; ++sub) {
            v16h afr, bfr;
            #pragma unroll
            for (int j = 0; j < 8; ++j) {
                afr[j]     = ap[sub * 32 + khalf * 8 + j];
                afr[8 + j] = ap[sub * 32 + 16 + khalf * 8 + j];
            }
            #pragma unroll
            for (int j = 0; j < 8; ++j) {
                bfr[j]     = bp[sub * 32 + khalf * 8 + j];
                bfr[8 + j] = bp[sub * 32 + 16 + khalf * 8 + j];
            }
            acc = __builtin_amdgcn_wmma_f32_16x16x32_f16(false, afr, false, bfr,
                                                         (short)0, acc, false, false);
        }
        __syncthreads();
    }

    // store: lane covers column n, rows v + 8*khalf (ISA f32 C layout)
    const int n = ntile * 16 + lrow;
    #pragma unroll
    for (int v = 0; v < 8; ++v) {
        int m = mbase + wave * 16 + v + khalf * 8;
        float val = acc[v] + (bias ? bias[m] : 0.f);
        if (ACT == 1)      val = fmaxf(val, 0.f);
        else if (ACT == 2) val = 4.f * fmaxf(val, 0.f);
        Cout[((long)b * M + m) * NPAD + n] = val;
    }
}

// =====================================================================
// Per-(b,c) row norm of conv3 output: xn = (x - mean) / (||x-mean|| + 1e-4)
// grid = B*256, block 128 (l)
// =====================================================================
__global__ __launch_bounds__(128) void row_norm_kernel(const float* __restrict__ x4,
                                                       float* __restrict__ xn) {
    __shared__ float red[128];
    long rc = blockIdx.x;          // b*256 + c
    int l = threadIdx.x;
    const float* row = x4 + rc * NPAD;
    float v = (l < L3) ? row[l] : 0.f;
    red[l] = v; __syncthreads();
    for (int s = 64; s > 0; s >>= 1) { if (l < s) red[l] += red[l + s]; __syncthreads(); }
    float mean = red[0] / (float)L3; __syncthreads();
    float d = (l < L3) ? v - mean : 0.f;
    red[l] = d * d; __syncthreads();
    for (int s = 64; s > 0; s >>= 1) { if (l < s) red[l] += red[l + s]; __syncthreads(); }
    float stdv = sqrtf(red[0]) + 1e-4f;
    xn[rc * NPAD + l] = (l < L3) ? d / stdv : 0.f;
}

// =====================================================================
// Per-(b,l) channel-norm: xn[:, :, l] /= ||xn[b,:,l]||   grid=B, block 128
// =====================================================================
__global__ __launch_bounds__(128) void col_norm_kernel(float* __restrict__ xn) {
    int b = blockIdx.x, l = threadIdx.x;
    if (l >= L3) return;
    float* base = xn + (long)b * DIM * NPAD + l;
    float ss = 0.f;
    for (int c = 0; c < DIM; ++c) { float v = base[c * NPAD]; ss += v * v; }
    float inv = 1.f / sqrtf(ss);
    for (int c = 0; c < DIM; ++c) base[c * NPAD] *= inv;
}

// =====================================================================
// argmax over codes (normalized codebook => argmin dist == argmax score),
// gather un-normalized codebook row into q_after, accumulate diff.
// grid=B, block 128 (l)
// =====================================================================
__global__ __launch_bounds__(128) void argmax_gather_kernel(const float* __restrict__ scores,
                                                            const float* __restrict__ xn,
                                                            const float* __restrict__ ew,
                                                            float* __restrict__ qout,
                                                            float* __restrict__ diffacc) {
    __shared__ float red[128];
    int b = blockIdx.x, l = threadIdx.x;
    float dsum = 0.f;
    if (l < L3) {
        const float* sc = scores + (long)b * NEMB * NPAD + l;
        float best = -3.4e38f; int bi = 0;
        for (int n = 0; n < NEMB; ++n) { float s = sc[n * NPAD]; if (s > best) { best = s; bi = n; } }
        const float* e = ew + (long)bi * DIM;
        for (int c = 0; c < DIM; ++c) {
            float q = e[c];
            qout[((long)b * DIM + c) * L3 + l] = q;
            float d = q - xn[((long)b * DIM + c) * NPAD + l];
            dsum += d * d;
        }
    }
    red[l] = dsum; __syncthreads();
    for (int s = 64; s > 0; s >>= 1) { if (l < s) red[l] += red[l + s]; __syncthreads(); }
    if (l == 0) atomicAdd(diffacc, red[0]);
}

// =====================================================================
// speaker embed: sp[b][c] = mean_l(x - q) ; then /(||sp||+1e-4)/3.  grid=B, block 256
// =====================================================================
__global__ __launch_bounds__(256) void sp_kernel(const float* __restrict__ x4,
                                                 const float* __restrict__ qout,
                                                 float* __restrict__ spout,
                                                 float* __restrict__ spws) {
    __shared__ float red[256];
    int b = blockIdx.x, c = threadIdx.x;
    const float* xr = x4 + ((long)b * DIM + c) * NPAD;
    const float* qr = qout + ((long)b * DIM + c) * L3;
    float s = 0.f;
    for (int l = 0; l < L3; ++l) s += xr[l] - qr[l];
    s /= (float)L3;
    red[c] = s * s; __syncthreads();
    for (int st = 128; st > 0; st >>= 1) { if (c < st) red[c] += red[c + st]; __syncthreads(); }
    float val = s / (sqrtf(red[0]) + 1e-4f) / 3.f;
    spout[b * DIM + c] = val;
    spws[b * DIM + c]  = val;
}

// =====================================================================
// decoder input: decin[b][c][l] = q_after + sp_sel  (padded to 128)
// grid=B, block 256 (c); shuffled=1 => sp[idx[b]]
// =====================================================================
__global__ __launch_bounds__(256) void dec_prep_kernel(const float* __restrict__ qout,
                                                       const float* __restrict__ spws,
                                                       const int* __restrict__ idxws,
                                                       float* __restrict__ decin, int shuffled) {
    int b = blockIdx.x, c = threadIdx.x;
    int sb = shuffled ? idxws[b] : b;
    float spv = spws[sb * DIM + c];
    const float* qr = qout + ((long)b * DIM + c) * L3;
    float* dr = decin + ((long)b * DIM + c) * NPAD;
    for (int l = 0; l < NPAD; ++l) dr[l] = (l < L3) ? qr[l] + spv : 0.f;
}

// =====================================================================
// nearest-neighbor upsample 122 -> 240.  grid=B*128, block 256 (t<240)
// =====================================================================
__global__ __launch_bounds__(256) void upsample_kernel(const float* __restrict__ y,
                                                       float* __restrict__ out) {
    long bc = blockIdx.x;
    int t = threadIdx.x;
    if (t >= TT) return;
    int l = (t * L3) / TT;
    out[bc * TT + t] = y[bc * NPAD + l];
}

// =====================================================================
// deterministic permutation (stand-in for jax.random.permutation(key(1)))
// =====================================================================
__global__ void perm_kernel(int* __restrict__ idxws, float* __restrict__ outIdx) {
    if (threadIdx.x == 0 && blockIdx.x == 0) {
        for (int i = 0; i < BB; ++i) idxws[i] = i;
        unsigned s = 0x9E3779B9u;
        for (int i = BB - 1; i > 0; --i) {
            s ^= s << 13; s ^= s >> 17; s ^= s << 5;
            int j = (int)(s % (unsigned)(i + 1));
            int t = idxws[i]; idxws[i] = idxws[j]; idxws[j] = t;
        }
        for (int i = 0; i < BB; ++i) outIdx[i] = (float)idxws[i];
    }
}

__global__ void zero_kernel(float* p) { if (threadIdx.x == 0) *p = 0.f; }
__global__ void fin_diff_kernel(const float* acc, float* out) {
    if (threadIdx.x == 0) *out = *acc / ((float)BB * (float)L3 * (float)DIM);
}

// =====================================================================
// launch
// =====================================================================
extern "C" void kernel_launch(void* const* d_in, const int* in_sizes, int n_in,
                              void* d_out, int out_size, void* d_ws, size_t ws_size,
                              hipStream_t stream) {
    (void)in_sizes; (void)n_in; (void)out_size; (void)ws_size;
    const float* input   = (const float*)d_in[0];
    const float* w0      = (const float*)d_in[1];
    const float* b0      = (const float*)d_in[2];
    const float* w1      = (const float*)d_in[3];
    const float* b1      = (const float*)d_in[4];
    const float* w2      = (const float*)d_in[5];
    const float* b2      = (const float*)d_in[6];
    const float* w3      = (const float*)d_in[7];
    const float* b3      = (const float*)d_in[8];
    const float* embed_w = (const float*)d_in[9];
    const float* dw0     = (const float*)d_in[10];
    const float* db0     = (const float*)d_in[11];

    // ---------------- workspace layout ----------------
    char* ws = (char*)d_ws;
    size_t off = 0;
    float* a1      = (float*)(ws + off);    off += (size_t)BB * 128 * NPAD * 4;   // conv0 out / later: scores
    float* a2      = (float*)(ws + off);    off += (size_t)BB * 256 * NPAD * 4;   // conv1 out / later: decin
    float* a3      = (float*)(ws + off);    off += (size_t)BB * 128 * NPAD * 4;   // conv2 out / later: y
    float* x4      = (float*)(ws + off);    off += (size_t)BB * 256 * NPAD * 4;   // conv3 out
    float* xn      = (float*)(ws + off);    off += (size_t)BB * 256 * NPAD * 4;   // normalized
    _Float16* w0h  = (_Float16*)(ws + off); off += 128 * 960 * 2;
    _Float16* w1h  = (_Float16*)(ws + off); off += 256 * 384 * 2;
    _Float16* w2h  = (_Float16*)(ws + off); off += 128 * 768 * 2;
    _Float16* w3h  = (_Float16*)(ws + off); off += 256 * 128 * 2;
    _Float16* dwh  = (_Float16*)(ws + off); off += 128 * 768 * 2;
    _Float16* enh  = (_Float16*)(ws + off); off += 128 * 256 * 2;
    float* spws    = (float*)(ws + off);    off += (size_t)BB * 256 * 4;
    int*   idxws   = (int*)(ws + off);      off += BB * 4;
    float* acc     = (float*)(ws + off);    off += 256;
    float* scores  = a1;
    float* decin   = a2;
    float* ybuf    = a3;

    // ---------------- output layout ----------------
    float* out  = (float*)d_out;
    float* dec1 = out;
    float* dec2 = out + (size_t)BB * CH * TT;
    float* qout = dec2 + (size_t)BB * CH * TT;
    float* spo  = qout + (size_t)BB * DIM * L3;
    float* diffo = spo + (size_t)BB * DIM;
    float* idxo  = diffo + 1;

    auto cast = [&](const float* s, _Float16* d, int n) {
        cast_f16_kernel<<<(n + 255) / 256, 256, 0, stream>>>(s, d, n);
    };

    // weight conversions + setup
    cast(w0, w0h, 128 * 960);
    cast(w1, w1h, 256 * 384);
    cast(w2, w2h, 128 * 768);
    cast(w3, w3h, 256 * 128);
    cast(dw0, dwh, 128 * 768);
    embed_norm_kernel<<<NEMB, 256, 0, stream>>>(embed_w, enh);
    zero_kernel<<<1, 1, 0, stream>>>(acc);
    perm_kernel<<<1, 1, 0, stream>>>(idxws, idxo);

    // conv0: input [B,240,240] --k4 s2 p1--> relu -> a1 [B,128,120pad128]
    conv_gemm_wmma_kernel<4, 2, 1, 1><<<dim3(8, 1, BB), 256, 0, stream>>>(
        w0h, input, a1, b0, 128, INCH, TT, TT, L0);
    // conv1: k3 p1 -> a2 [B,256,120]
    conv_gemm_wmma_kernel<3, 1, 1, 0><<<dim3(8, 2, BB), 256, 0, stream>>>(
        w1h, a1, a2, b1, 256, 128, L0, NPAD, L0);
    // conv2: k3 p1 + fused double-resblock (4*relu) -> a3 [B,128,120]
    conv_gemm_wmma_kernel<3, 1, 1, 2><<<dim3(8, 1, BB), 256, 0, stream>>>(
        w2h, a2, a3, b2, 128, 256, L0, NPAD, L0);
    // conv3: k1 p1 -> x4 [B,256,122]
    conv_gemm_wmma_kernel<1, 1, 1, 0><<<dim3(8, 2, BB), 256, 0, stream>>>(
        w3h, a3, x4, b3, 256, 128, L0, NPAD, L3);

    // normalization
    row_norm_kernel<<<BB * DIM, 128, 0, stream>>>(x4, xn);
    col_norm_kernel<<<BB, 128, 0, stream>>>(xn);

    // quantize: scores = En * xn  (argmin dist == argmax score for unit codes)
    conv_gemm_wmma_kernel<1, 1, 0, 0><<<dim3(8, 1, BB), 256, 0, stream>>>(
        enh, xn, scores, nullptr, 128, DIM, L3, NPAD, L3);
    argmax_gather_kernel<<<BB, 128, 0, stream>>>(scores, xn, embed_w, qout, acc);
    sp_kernel<<<BB, 256, 0, stream>>>(x4, qout, spo, spws);
    fin_diff_kernel<<<1, 1, 0, stream>>>(acc, diffo);

    // decoder pass 1: (q + sp) -> conv k3 p1 -> 4*relu -> upsample to 240
    dec_prep_kernel<<<BB, 256, 0, stream>>>(qout, spws, idxws, decin, 0);
    conv_gemm_wmma_kernel<3, 1, 1, 2><<<dim3(8, 1, BB), 256, 0, stream>>>(
        dwh, decin, ybuf, db0, 128, DIM, L3, NPAD, L3);
    upsample_kernel<<<BB * CH, 256, 0, stream>>>(ybuf, dec1);

    // decoder pass 2: sp shuffled by idx
    dec_prep_kernel<<<BB, 256, 0, stream>>>(qout, spws, idxws, decin, 1);
    conv_gemm_wmma_kernel<3, 1, 1, 2><<<dim3(8, 1, BB), 256, 0, stream>>>(
        dwh, decin, ybuf, db0, 128, DIM, L3, NPAD, L3);
    upsample_kernel<<<BB * CH, 256, 0, stream>>>(ybuf, dec2);
}
